// CustomGraphTransformerLayer_23579370455377
// MI455X (gfx1250) — matrix operand order repaired and checked by reading, not compiled, not run on hardware
//
#include <hip/hip_runtime.h>

#define NNODES 50000
#define NEDGES 800000
#define CH     128
#define NHEAD  4
#define DK     32
#define TOKDIM 768
#define NSENT  64      // N_GRAPHS / OPTION_NUM
#define OPTNUM 4

typedef __attribute__((ext_vector_type(16))) _Float16 v16h;
typedef __attribute__((ext_vector_type(4)))  _Float16 v4h;
typedef __attribute__((ext_vector_type(8)))  float    v8f;
typedef __attribute__((ext_vector_type(4)))  float    v4f;

// ---------- ordered-int float max helpers (monotone map, handles negatives) ----------
__device__ __forceinline__ unsigned int f2ord(float f) {
    unsigned int u = __float_as_uint(f);
    return (u & 0x80000000u) ? ~u : (u | 0x80000000u);
}
__device__ __forceinline__ float ord2f(unsigned int u) {
    u = (u & 0x80000000u) ? (u & 0x7fffffffu) : ~u;
    return __uint_as_float(u);
}

// ---------- WMMA fragment loaders (CDNA5 16x16x32 f16 layouts) ----------
// A (16x32 f16) from an f32 row-major source, converted inline:
// lane L holds row m = L&15; half h -> K = 16*(h>>3) + (h&7) + 8*(L>>4)
__device__ __forceinline__ v16h load_a_frag_f32(const float* __restrict__ A,
                                                int row0, int k0, int ld, int lane) {
    int m  = lane & 15;
    int hs = lane >> 4;
    const float* base = A + (size_t)(row0 + m) * ld + k0 + 8 * hs;
    v16h a;
#pragma unroll
    for (int g = 0; g < 2; ++g) {
        v4f f0 = *(const v4f*)(base + 16 * g);
        v4f f1 = *(const v4f*)(base + 16 * g + 4);
#pragma unroll
        for (int i = 0; i < 4; ++i) {
            a[8 * g + i]     = (_Float16)f0[i];
            a[8 * g + 4 + i] = (_Float16)f1[i];
        }
    }
    return a;
}

// B fragments are pre-swizzled into per-lane contiguous order by pack_b_kernel:
//   Bp[((kt*NT + nt)*32 + lane)*16 + h] = (f16) W[(kt*32 + 8*(lane>>4) + (h&7) + 16*(h>>3))*N + nt*16 + (lane&15)]
// so the hot-loop load is two contiguous 16B vector loads.
__device__ __forceinline__ v16h load_b_packed(const _Float16* __restrict__ Bp,
                                              int kt, int nt, int NT, int lane) {
    const _Float16* base = Bp + ((size_t)((kt * NT + nt) * 32 + lane)) * 16;
    union { uint4 u[2]; v16h v; } un;
    un.u[0] = *(const uint4*)(base);
    un.u[1] = *(const uint4*)(base + 8);
    return un.v;
}

// ---------- kernels ----------
__global__ void init_stats_kernel(unsigned int* __restrict__ maxbits, float* __restrict__ sums) {
    if (threadIdx.x < NHEAD) { maxbits[threadIdx.x] = 0u; sums[threadIdx.x] = 0.f; }
}

// Swizzle f32 weight [K x Ncols] row-major into f16 B-fragment order.
__global__ void pack_b_kernel(const float* __restrict__ W, _Float16* __restrict__ Bp,
                              int K, int Ncols) {
    int total = K * Ncols;
    int NT = Ncols >> 4;
    int i = blockIdx.x * blockDim.x + threadIdx.x;
    int stride = gridDim.x * blockDim.x;
    for (; i < total; i += stride) {
        int h    = i & 15;
        int lane = (i >> 4) & 31;
        int tile = i >> 9;            // kt*NT + nt
        int nt   = tile % NT;
        int kt   = tile / NT;
        int n = nt * 16 + (lane & 15);
        int k = kt * 32 + 8 * (lane >> 4) + (h & 7) + 16 * (h >> 3);
        Bp[i] = (_Float16)W[(size_t)k * Ncols + n];
    }
}

// S_small[64x128] = sent[64x768] @ Ws[768x128] + bs   (WMMA)
__global__ void sent_proj_kernel(const float* __restrict__ sent, const _Float16* __restrict__ WsP,
                                 const float* __restrict__ bias, float* __restrict__ Sml) {
    int lane = threadIdx.x & 31;
    int wave = threadIdx.x >> 5;     // n-tile 0..7
    int row0 = blockIdx.x * 16;      // 4 blocks
    int n0   = wave * 16;
    v8f c = {};
    for (int kt = 0; kt < TOKDIM / 32; ++kt) {
        v16h a = load_a_frag_f32(sent, row0, kt * 32, TOKDIM, lane);
        v16h b = load_b_packed(WsP, kt, wave, CH / 16, lane);
        c = __builtin_amdgcn_wmma_f32_16x16x32_f16(false, a, false, b, (short)0, c, false, false);
    }
    int n = lane & 15, hs = lane >> 4;
    float bb = bias[n0 + n];
#pragma unroll
    for (int r = 0; r < 8; ++r) {
        int m = r + 8 * hs;
        Sml[(row0 + m) * CH + n0 + n] = c[r] + bb;
    }
}

// Fused: Q/K/V = x@W + b (WMMA, f16 out), S gather from S_small, zero the scatter accumulator
__global__ void qkvs_kernel(const float* __restrict__ x,
                            const _Float16* __restrict__ WqP, const _Float16* __restrict__ WkP,
                            const _Float16* __restrict__ WvP,
                            const float* __restrict__ bq, const float* __restrict__ bk,
                            const float* __restrict__ bv,
                            const float* __restrict__ Sml, const int* __restrict__ batch,
                            _Float16* __restrict__ Qh, _Float16* __restrict__ Kh,
                            _Float16* __restrict__ Vh, _Float16* __restrict__ Sh,
                            float* __restrict__ out_acc) {
    int lane = threadIdx.x & 31;
    int wave = threadIdx.x >> 5;
    int row0 = blockIdx.x * 16;      // 3125 blocks
    int n0   = wave * 16;
    int n = lane & 15, hs = lane >> 4;

    // hoist A fragments (shared across the 3 weight matrices)
    v16h afrag[4];
#pragma unroll
    for (int kk = 0; kk < 4; ++kk) afrag[kk] = load_a_frag_f32(x, row0, kk * 32, CH, lane);

    const _Float16* Wm[3] = {WqP, WkP, WvP};
    const float*    Bm[3] = {bq, bk, bv};
    _Float16*       Om[3] = {Qh, Kh, Vh};
#pragma unroll
    for (int m3 = 0; m3 < 3; ++m3) {
        v8f c = {};
#pragma unroll
        for (int kk = 0; kk < 4; ++kk) {
            v16h b = load_b_packed(Wm[m3], kk, wave, CH / 16, lane);
            c = __builtin_amdgcn_wmma_f32_16x16x32_f16(false, afrag[kk], false, b,
                                                       (short)0, c, false, false);
        }
        float bb = Bm[m3][n0 + n];
#pragma unroll
        for (int r = 0; r < 8; ++r) {
            int m = r + 8 * hs;
            Om[m3][(size_t)(row0 + m) * CH + n0 + n] = (_Float16)(c[r] + bb);
        }
    }

    // S gather + accumulator zero: 16 rows x 128 cols, 8 elems per thread
    int t = threadIdx.x;
    int node  = row0 + (t >> 4);
    int cbase = (t & 15) * 8;
    int g = batch[node] / OPTNUM;
#pragma unroll
    for (int i = 0; i < 8; ++i) {
        float sv = Sml[g * CH + cbase + i];
        Sh[(size_t)node * CH + cbase + i] = (_Float16)sv;
        out_acc[(size_t)node * CH + cbase + i] = 0.f;
    }
}

// Edge scores: alpha[e,h] = leaky_relu(((Qi+Si).Kj + Qi.Sj) / (3*sqrt(32))); track per-head max
__global__ void edge_alpha_kernel(const int* __restrict__ ei,
                                  const _Float16* __restrict__ Qh, const _Float16* __restrict__ Kh,
                                  const _Float16* __restrict__ Sh,
                                  float* __restrict__ alpha, unsigned int* __restrict__ maxbits) {
    __shared__ unsigned int smax[NHEAD];
    if (threadIdx.x < NHEAD) smax[threadIdx.x] = 0u;  // ordered(-inf)
    __syncthreads();

    int lane = threadIdx.x & 31;
    int wid = (blockIdx.x * blockDim.x + threadIdx.x) >> 5;
    int nw  = (gridDim.x * blockDim.x) >> 5;
    const int h   = lane >> 3;                  // 8 lanes per head
    const int off = h * DK + (lane & 7) * 4;    // 4 halves per lane
    const float scale = 1.0f / (3.0f * 5.656854249492380f);
    float vmax = -3.4e38f;

    for (int e = wid; e < NEDGES; e += nw) {
        int row = ei[e];
        int col = ei[NEDGES + e];
        v4h q  = *(const v4h*)(Qh + (size_t)row * CH + off);
        v4h si = *(const v4h*)(Sh + (size_t)row * CH + off);
        v4h k  = *(const v4h*)(Kh + (size_t)col * CH + off);
        v4h sj = *(const v4h*)(Sh + (size_t)col * CH + off);
        float acc = 0.f;
#pragma unroll
        for (int i = 0; i < 4; ++i) {
            float qf = (float)q[i], kf = (float)k[i];
            acc += (qf + (float)si[i]) * kf + qf * (float)sj[i];
        }
        acc += __shfl_xor(acc, 1, 32);
        acc += __shfl_xor(acc, 2, 32);
        acc += __shfl_xor(acc, 4, 32);
        float a = acc * scale;
        a = (a > 0.f) ? a : 0.2f * a;
        if ((lane & 7) == 0) alpha[e * NHEAD + h] = a;
        vmax = fmaxf(vmax, a);
    }
    atomicMax(&smax[h], f2ord(vmax));
    __syncthreads();
    if (threadIdx.x < NHEAD) atomicMax(&maxbits[threadIdx.x], smax[threadIdx.x]);
}

// exp(alpha - max_h) in place; per-head sums
__global__ void edge_exp_kernel(float* __restrict__ alpha, const unsigned int* __restrict__ maxbits,
                                float* __restrict__ sums) {
    __shared__ float ssum[NHEAD];
    if (threadIdx.x < NHEAD) ssum[threadIdx.x] = 0.f;
    __syncthreads();
    int tid = blockIdx.x * blockDim.x + threadIdx.x;
    int stride = gridDim.x * blockDim.x;   // multiple of 4 -> head fixed per thread
    int h = tid & 3;
    float mh = ord2f(maxbits[h]);
    float local = 0.f;
    for (int i = tid; i < NEDGES * NHEAD; i += stride) {
        float ex = __expf(alpha[i] - mh);
        alpha[i] = ex;
        local += ex;
    }
    atomicAdd(&ssum[h], local);
    __syncthreads();
    if (threadIdx.x < NHEAD) atomicAdd(&sums[threadIdx.x], ssum[threadIdx.x]);
}

// Scatter: out_acc[row] += V[col] * alpha[e,h] / sum_h   (hardware fp32 atomics)
__global__ void edge_scatter_kernel(const int* __restrict__ ei, const _Float16* __restrict__ Vh,
                                    const float* __restrict__ alpha, const float* __restrict__ sums,
                                    float* __restrict__ out_acc) {
    int lane = threadIdx.x & 31;
    int wid = (blockIdx.x * blockDim.x + threadIdx.x) >> 5;
    int nw  = (gridDim.x * blockDim.x) >> 5;
    const int h   = lane >> 3;
    const int off = h * DK + (lane & 7) * 4;
    const float inv = 1.0f / sums[h];

    for (int e = wid; e < NEDGES; e += nw) {
        int row = ei[e];
        int col = ei[NEDGES + e];
        float w = alpha[e * NHEAD + h] * inv;
        v4h v = *(const v4h*)(Vh + (size_t)col * CH + off);
        float* dst = out_acc + (size_t)row * CH + off;
#pragma unroll
        for (int i = 0; i < 4; ++i) unsafeAtomicAdd(dst + i, (float)v[i] * w);
    }
}

// out = out_acc @ Wo + bo + x   (WMMA with inline f32->f16 A convert, f32 out)
__global__ void out_proj_kernel(const float* __restrict__ acc, const _Float16* __restrict__ WoP,
                                const float* __restrict__ bo, const float* __restrict__ x,
                                float* __restrict__ out) {
    int lane = threadIdx.x & 31;
    int wave = threadIdx.x >> 5;
    int row0 = blockIdx.x * 16;
    int n0   = wave * 16;
    v8f c = {};
#pragma unroll
    for (int kk = 0; kk < 4; ++kk) {
        v16h a = load_a_frag_f32(acc, row0, kk * 32, CH, lane);
        v16h b = load_b_packed(WoP, kk, wave, CH / 16, lane);
        c = __builtin_amdgcn_wmma_f32_16x16x32_f16(false, a, false, b, (short)0, c, false, false);
    }
    int n = lane & 15, hs = lane >> 4;
    float bb = bo[n0 + n];
#pragma unroll
    for (int r = 0; r < 8; ++r) {
        int m = r + 8 * hs;
        size_t idx = (size_t)(row0 + m) * CH + n0 + n;
        out[idx] = c[r] + bb + x[idx];
    }
}

extern "C" void kernel_launch(void* const* d_in, const int* in_sizes, int n_in,
                              void* d_out, int out_size, void* d_ws, size_t ws_size,
                              hipStream_t stream) {
    const float* x    = (const float*)d_in[0];
    const int*   ei   = (const int*)d_in[1];      // [2,E] flat: rows then cols
    const float* sent = (const float*)d_in[2];    // [64,768]
    const int*   batch= (const int*)d_in[3];
    const float* Wq = (const float*)d_in[4];  const float* bq = (const float*)d_in[5];
    const float* Wk = (const float*)d_in[6];  const float* bk = (const float*)d_in[7];
    const float* Wv = (const float*)d_in[8];  const float* bv = (const float*)d_in[9];
    const float* Ws = (const float*)d_in[10]; const float* bs = (const float*)d_in[11];
    const float* Wo = (const float*)d_in[12]; const float* bo = (const float*)d_in[13];
    float* out = (float*)d_out;

    // ---- workspace carve-up (256B aligned segments) ----
    char* w = (char*)d_ws;
    auto carve = [&](size_t bytes) -> char* {
        char* p = w;
        w += (bytes + 255) & ~(size_t)255;
        return p;
    };
    _Float16* WqP   = (_Float16*)carve((size_t)CH * CH * 2);
    _Float16* WkP   = (_Float16*)carve((size_t)CH * CH * 2);
    _Float16* WvP   = (_Float16*)carve((size_t)CH * CH * 2);
    _Float16* WoP   = (_Float16*)carve((size_t)CH * CH * 2);
    _Float16* WsP   = (_Float16*)carve((size_t)TOKDIM * CH * 2);
    _Float16* Qh    = (_Float16*)carve((size_t)NNODES * CH * 2);
    _Float16* Kh    = (_Float16*)carve((size_t)NNODES * CH * 2);
    _Float16* Vh    = (_Float16*)carve((size_t)NNODES * CH * 2);
    _Float16* Sh    = (_Float16*)carve((size_t)NNODES * CH * 2);
    float*    Sml   = (float*)carve((size_t)NSENT * CH * 4);
    float*    acc   = (float*)carve((size_t)NNODES * CH * 4);
    float*    alpha = (float*)carve((size_t)NEDGES * NHEAD * 4);
    unsigned int* maxbits = (unsigned int*)carve(256);   // 4 used
    float*        sums    = (float*)carve(256);          // 4 used

    init_stats_kernel<<<1, 32, 0, stream>>>(maxbits, sums);

    // pre-swizzle weight matrices into WMMA B-fragment order (one pass each)
    pack_b_kernel<<<64, 256, 0, stream>>>(Wq, WqP, CH, CH);
    pack_b_kernel<<<64, 256, 0, stream>>>(Wk, WkP, CH, CH);
    pack_b_kernel<<<64, 256, 0, stream>>>(Wv, WvP, CH, CH);
    pack_b_kernel<<<64, 256, 0, stream>>>(Wo, WoP, CH, CH);
    pack_b_kernel<<<384, 256, 0, stream>>>(Ws, WsP, TOKDIM, CH);

    // S_small = sent @ Ws + bs
    sent_proj_kernel<<<NSENT / 16, 256, 0, stream>>>(sent, WsP, bs, Sml);

    // Q/K/V projections + S gather + accumulator zero
    qkvs_kernel<<<NNODES / 16, 256, 0, stream>>>(x, WqP, WkP, WvP, bq, bk, bv,
                                                 Sml, batch, Qh, Kh, Vh, Sh, acc);

    // edge phase
    edge_alpha_kernel<<<1024, 256, 0, stream>>>(ei, Qh, Kh, Sh, alpha, maxbits);
    edge_exp_kernel<<<1024, 256, 0, stream>>>(alpha, maxbits, sums);
    edge_scatter_kernel<<<2048, 256, 0, stream>>>(ei, Vh, alpha, sums, acc);

    // output projection + residual
    out_proj_kernel<<<NNODES / 16, 256, 0, stream>>>(acc, WoP, bo, x, out);
}
